// DiTBlock_53850299957720
// MI455X (gfx1250) — compile-verified, matrix operand order checked by reading, not run on hardware
//
#include <hip/hip_runtime.h>
#include <math.h>

// ---------------------------------------------------------------------------
// DiT block forward for MI455X (gfx1250, wave32, WMMA bf16 16x16x32 path).
// Strategy: all large GEMMs run as bf16 WMMA with f32 accumulation; LN /
// softmax / GELU / residuals / modulation stay f32.
// ---------------------------------------------------------------------------

typedef __attribute__((ext_vector_type(16))) __bf16 v16bf;
typedef __attribute__((ext_vector_type(8)))  float  v8f;

#define BB 8
#define SS 1024
#define DD 1024
#define FF 4096
#define HH 16
#define DH 64
#define MM (BB * SS)     // 8192 rows

__device__ __forceinline__ unsigned short f2bf(float f) {
    unsigned int u = __float_as_uint(f);
    unsigned int r = (u + 0x7FFFu + ((u >> 16) & 1u)) >> 16;
    return (unsigned short)r;
}

// Load a 16x32 bf16 fragment (A-style layout per CDNA5 ISA 7.12.2) from LDS.
// Lane l: M = l&15, K-halfgroup base = (l>>4)*8; VGPR v<4 -> K = base+2v,
// VGPR v>=4 -> K = 16+base+2(v-4). Each VGPR = two consecutive bf16 = one b32.
// B operand (32x16 KxN) uses the mirrored layout with N in place of M, so the
// same loader works when the B tile is staged n-major in LDS.
__device__ __forceinline__ v16bf load_frag16(const unsigned short* base,
                                             int row0, int k0, int ld) {
    const int lane = threadIdx.x & 31;
    const int r  = row0 + (lane & 15);
    const int kb = (lane >> 4) << 3;
    const unsigned short* p = base + r * ld + k0 + kb;
    union { v16bf v; unsigned int u[8]; } f;
#pragma unroll
    for (int i = 0; i < 4; ++i)
        f.u[i] = *reinterpret_cast<const unsigned int*>(p + 2 * i);
#pragma unroll
    for (int i = 0; i < 4; ++i)
        f.u[4 + i] = *reinterpret_cast<const unsigned int*>(p + 16 + 2 * i);
    return f.v;
}

// ---------------------------------------------------------------------------
// small elementwise / setup kernels
// ---------------------------------------------------------------------------

__global__ void f32_to_bf16_kernel(const float* __restrict__ src,
                                   unsigned short* __restrict__ dst, int n) {
    int i = blockIdx.x * 256 + threadIdx.x;
    if (i < n) dst[i] = f2bf(src[i]);
}

__global__ void silu_kernel(const float* __restrict__ c,
                            float* __restrict__ cact, int n) {
    int i = blockIdx.x * 256 + threadIdx.x;
    if (i < n) {
        float x = c[i];
        cact[i] = x / (1.f + __expf(-x));
    }
}

// cmod = silu(c) @ w_mod + b_mod   (8 x 6144, tiny: naive f32)
__global__ __launch_bounds__(256) void cmod_kernel(
    const float* __restrict__ cact, const float* __restrict__ w_mod,
    const float* __restrict__ b_mod, float* __restrict__ cmod_out,
    float* __restrict__ shift_out, float* __restrict__ scale_out) {
    int col = blockIdx.x * 256 + threadIdx.x;   // 0..6143
    int b   = blockIdx.y;                       // 0..7
    const float* ca = cact + b * DD;
    float acc = b_mod[col];
    for (int kk = 0; kk < DD; ++kk)
        acc = fmaf(ca[kk], w_mod[(size_t)kk * (6 * DD) + col], acc);
    cmod_out[(size_t)b * (6 * DD) + col] = acc;
    if (col < DD)            shift_out[b * DD + col]        = acc;
    else if (col < 2 * DD)   scale_out[b * DD + (col - DD)] = acc;
}

// LayerNorm (eps 1e-6, biased var) + modulate; optional f32 outputs + bf16 copy
__global__ __launch_bounds__(256) void ln_mod_kernel(
    const float* __restrict__ x, const float* __restrict__ cmod,
    int shift_off, int scale_off,
    float* __restrict__ xn_out, float* __restrict__ xm_out,
    unsigned short* __restrict__ xm_bf) {
    const int row = blockIdx.x;
    const int tid = threadIdx.x;
    const int b   = row >> 10;  // S = 1024
    const float* xr = x + (size_t)row * DD;
    float v[4], s = 0.f, sq = 0.f;
#pragma unroll
    for (int j = 0; j < 4; ++j) {
        v[j] = xr[tid + 256 * j];
        s += v[j];
        sq = fmaf(v[j], v[j], sq);
    }
#pragma unroll
    for (int off = 16; off >= 1; off >>= 1) {
        s  += __shfl_xor(s, off, 32);
        sq += __shfl_xor(sq, off, 32);
    }
    __shared__ float rs[8], rq[8];
    if ((tid & 31) == 0) { rs[tid >> 5] = s; rq[tid >> 5] = sq; }
    __syncthreads();
    s = 0.f; sq = 0.f;
#pragma unroll
    for (int w = 0; w < 8; ++w) { s += rs[w]; sq += rq[w]; }
    const float mean = s * (1.f / DD);
    const float var  = sq * (1.f / DD) - mean * mean;
    const float rstd = rsqrtf(var + 1e-6f);
    const float* shp = cmod + (size_t)b * (6 * DD) + shift_off;
    const float* scp = cmod + (size_t)b * (6 * DD) + scale_off;
#pragma unroll
    for (int j = 0; j < 4; ++j) {
        int col  = tid + 256 * j;
        float xn = (v[j] - mean) * rstd;
        float xm = fmaf(xn, 1.f + scp[col], shp[col]);
        if (xn_out) xn_out[(size_t)row * DD + col] = xn;
        if (xm_out) xm_out[(size_t)row * DD + col] = xm;
        xm_bf[(size_t)row * DD + col] = f2bf(xm);
    }
}

// ---------------------------------------------------------------------------
// Tiled bf16 WMMA GEMM: C[M,N] = A[M,K] @ B[K,N] (+ epilogue)
// Workgroup tile 128x64, 8 waves, each wave 16(M) x 64(N) = 4 accum tiles.
// EPI 0: outB = bf16((acc+bias)*scale)
// EPI 1: outF = res + gate[b, gate_off+col] * (acc+bias)      (f32)
// EPI 2: outB = bf16(gelu_exact(acc+bias))
// ---------------------------------------------------------------------------
template <int EPI>
__global__ __launch_bounds__(256) void gemm_bf16_kernel(
    const unsigned short* __restrict__ A, const unsigned short* __restrict__ Bw,
    const float* __restrict__ bias, const float* __restrict__ cmod, int gate_off,
    const float* __restrict__ res, float* __restrict__ outF,
    unsigned short* __restrict__ outB, int M, int N, int K, float scale) {
    __shared__ unsigned short As[128][36];   // row-major 128x32 (+pad)
    __shared__ unsigned short Bs[64][36];    // n-major 64x32 (+pad)
    const int tid  = threadIdx.x;
    const int wave = tid >> 5;
    const int lane = tid & 31;
    const int m0 = blockIdx.y * 128;
    const int n0 = blockIdx.x * 64;

    v8f acc[4];
#pragma unroll
    for (int nt = 0; nt < 4; ++nt)
#pragma unroll
        for (int r = 0; r < 8; ++r) acc[nt][r] = 0.f;

    for (int k0 = 0; k0 < K; k0 += 32) {
        __syncthreads();
        {   // A: 128x32 bf16, 16 elems/thread
            int row = tid >> 1;
            int cb  = (tid & 1) << 4;
            const unsigned short* gp = A + (size_t)(m0 + row) * K + k0 + cb;
            uint4 d0 = *(const uint4*)gp;
            uint4 d1 = *(const uint4*)(gp + 8);
            unsigned short* sp = &As[row][cb];
            *(uint2*)(sp + 0)  = make_uint2(d0.x, d0.y);
            *(uint2*)(sp + 4)  = make_uint2(d0.z, d0.w);
            *(uint2*)(sp + 8)  = make_uint2(d1.x, d1.y);
            *(uint2*)(sp + 12) = make_uint2(d1.z, d1.w);
        }
        {   // B: 32x64 -> transpose to Bs[n][k]
            int kk = tid >> 3;
            int nb = (tid & 7) << 3;
            const unsigned short* gp = Bw + (size_t)(k0 + kk) * N + n0 + nb;
            uint4 d = *(const uint4*)gp;
            const unsigned short* e = (const unsigned short*)&d;
#pragma unroll
            for (int j = 0; j < 8; ++j) Bs[nb + j][kk] = e[j];
        }
        if (k0 + 32 < K)  // hint next A tile toward the caches
            __builtin_prefetch(A + (size_t)(m0 + (tid >> 1)) * K + k0 + 32, 0, 1);
        __syncthreads();

        v16bf af = load_frag16(&As[0][0], wave * 16, 0, 36);
#pragma unroll
        for (int nt = 0; nt < 4; ++nt) {
            v16bf bf = load_frag16(&Bs[0][0], nt * 16, 0, 36);
            acc[nt] = __builtin_amdgcn_wmma_f32_16x16x32_bf16(
                false, af, false, bf, (short)0, acc[nt], false, false);
        }
    }

    const int colb = lane & 15;
    const int rh   = (lane >> 4) << 3;
#pragma unroll
    for (int nt = 0; nt < 4; ++nt) {
        int col  = n0 + nt * 16 + colb;
        float bv = bias[col];
#pragma unroll
        for (int r = 0; r < 8; ++r) {
            int row = m0 + wave * 16 + rh + r;
            float v = acc[nt][r] + bv;
            if (EPI == 0) {
                outB[(size_t)row * N + col] = f2bf(v * scale);
            } else if (EPI == 1) {
                int b  = row >> 10;  // S = 1024
                float g = cmod[(size_t)b * (6 * DD) + gate_off + col];
                outF[(size_t)row * N + col] =
                    fmaf(g, v, res[(size_t)row * N + col]);
            } else {
                float ge = 0.5f * v * (1.f + erff(v * 0.70710678118654752f));
                outB[(size_t)row * N + col] = f2bf(ge);
            }
        }
    }
}

// ---------------------------------------------------------------------------
// Flash attention: one (b,h) and 128 query rows per workgroup; 8 waves, each
// owning 16 query rows. Online softmax across key blocks of 32.
// q already carries the 1/Dh scale from the QKV GEMM epilogue.
// ---------------------------------------------------------------------------
__global__ __launch_bounds__(256) void attn_kernel(
    const unsigned short* __restrict__ q, const unsigned short* __restrict__ k,
    const unsigned short* __restrict__ v, unsigned short* __restrict__ y) {
    __shared__ unsigned short Qs[128][68];     // [qrow][dh]
    __shared__ unsigned short Ks[32][68];      // [key][dh]   (n-major for S=QK^T)
    __shared__ unsigned short Vt[64][36];      // [dh][key]   (n-major for O=PV)
    __shared__ unsigned short Ps[8][16][36];   // per-wave P tile [qrow][key]

    const int tid  = threadIdx.x;
    const int lane = tid & 31;
    const int wave = tid >> 5;
    const int bh = blockIdx.y;
    const int b  = bh >> 4;
    const int h  = bh & 15;
    const int q0 = blockIdx.x * 128;
    const size_t baseq = ((size_t)(b * SS + q0)) * DD + h * DH;

    {   // stage Q block: 128 x 64 bf16
        int row = tid >> 1;
        int cb  = (tid & 1) * 32;
        const unsigned short* gp = q + baseq + (size_t)row * DD + cb;
#pragma unroll
        for (int j = 0; j < 4; ++j) {
            uint4 d = *(const uint4*)(gp + 8 * j);
            unsigned short* sp = &Qs[row][cb + 8 * j];
            *(uint2*)(sp + 0) = make_uint2(d.x, d.y);
            *(uint2*)(sp + 4) = make_uint2(d.z, d.w);
        }
    }

    float m_i[8], l_i[8];
#pragma unroll
    for (int r = 0; r < 8; ++r) { m_i[r] = -1.0e30f; l_i[r] = 0.f; }
    v8f accO[4];
#pragma unroll
    for (int nt = 0; nt < 4; ++nt)
#pragma unroll
        for (int r = 0; r < 8; ++r) accO[nt][r] = 0.f;

    for (int kb = 0; kb < SS / 32; ++kb) {
        __syncthreads();   // protect Ks/Vt/Ps reuse (also covers Q staging)
        const size_t basek = ((size_t)(b * SS + kb * 32)) * DD + h * DH;
        {   // stage K (row = key) and V (transposed to [dh][key])
            int key = tid >> 3;
            int cb  = (tid & 7) * 8;
            uint4 dk = *(const uint4*)(k + basek + (size_t)key * DD + cb);
            unsigned short* sp = &Ks[key][cb];
            *(uint2*)(sp + 0) = make_uint2(dk.x, dk.y);
            *(uint2*)(sp + 4) = make_uint2(dk.z, dk.w);
            uint4 dv = *(const uint4*)(v + basek + (size_t)key * DD + cb);
            const unsigned short* e = (const unsigned short*)&dv;
#pragma unroll
            for (int j = 0; j < 8; ++j) Vt[cb + j][key] = e[j];
        }
        __syncthreads();

        // S tile: 16(q) x 32(keys), contraction over Dh=64 (two K=32 steps)
        v8f sc[2];
#pragma unroll
        for (int nt = 0; nt < 2; ++nt)
#pragma unroll
            for (int r = 0; r < 8; ++r) sc[nt][r] = 0.f;
        v16bf a0 = load_frag16(&Qs[0][0], wave * 16, 0, 68);
        v16bf a1 = load_frag16(&Qs[0][0], wave * 16, 32, 68);
#pragma unroll
        for (int nt = 0; nt < 2; ++nt) {
            v16bf b0 = load_frag16(&Ks[0][0], nt * 16, 0, 68);
            v16bf b1 = load_frag16(&Ks[0][0], nt * 16, 32, 68);
            sc[nt] = __builtin_amdgcn_wmma_f32_16x16x32_bf16(
                false, a0, false, b0, (short)0, sc[nt], false, false);
            sc[nt] = __builtin_amdgcn_wmma_f32_16x16x32_bf16(
                false, a1, false, b1, (short)0, sc[nt], false, false);
        }

        // online softmax; each 16-lane half holds one set of 8 rows
#pragma unroll
        for (int r = 0; r < 8; ++r) {
            float mx = fmaxf(sc[0][r], sc[1][r]);
#pragma unroll
            for (int off = 8; off >= 1; off >>= 1)
                mx = fmaxf(mx, __shfl_xor(mx, off, 16));
            float mn    = fmaxf(m_i[r], mx);
            float alpha = __expf(m_i[r] - mn);
            float p0 = __expf(sc[0][r] - mn);
            float p1 = __expf(sc[1][r] - mn);
            float sum = p0 + p1;
#pragma unroll
            for (int off = 8; off >= 1; off >>= 1)
                sum += __shfl_xor(sum, off, 16);
            l_i[r] = l_i[r] * alpha + sum;
            m_i[r] = mn;
            sc[0][r] = p0;
            sc[1][r] = p1;
#pragma unroll
            for (int nt = 0; nt < 4; ++nt) accO[nt][r] *= alpha;
        }

        {   // P tile -> per-wave LDS (C layout -> A-fragment layout)
            int rr = (lane >> 4) << 3;
#pragma unroll
            for (int r = 0; r < 8; ++r) {
                Ps[wave][rr + r][lane & 15]        = f2bf(sc[0][r]);
                Ps[wave][rr + r][16 + (lane & 15)] = f2bf(sc[1][r]);
            }
        }
        __syncthreads();

        // O += P @ V
        v16bf pa = load_frag16(&Ps[wave][0][0], 0, 0, 36);
#pragma unroll
        for (int nt = 0; nt < 4; ++nt) {
            v16bf bv = load_frag16(&Vt[0][0], nt * 16, 0, 36);
            accO[nt] = __builtin_amdgcn_wmma_f32_16x16x32_bf16(
                false, pa, false, bv, (short)0, accO[nt], false, false);
        }
    }

    {   // finalize: O /= l, store bf16 y[b, row, h*64 + dh]
        int rr = (lane >> 4) << 3;
        int cb = lane & 15;
#pragma unroll
        for (int r = 0; r < 8; ++r) {
            float inv = 1.0f / l_i[r];
            int row = q0 + wave * 16 + rr + r;
            size_t base = ((size_t)(b * SS + row)) * DD + h * DH;
#pragma unroll
            for (int nt = 0; nt < 4; ++nt)
                y[base + nt * 16 + cb] = f2bf(accO[nt][r] * inv);
        }
    }
}

// ---------------------------------------------------------------------------
// host-side launch
// ---------------------------------------------------------------------------

extern "C" void kernel_launch(void* const* d_in, const int* in_sizes, int n_in,
                              void* d_out, int out_size, void* d_ws, size_t ws_size,
                              hipStream_t stream) {
    (void)in_sizes; (void)n_in; (void)out_size; (void)ws_size;

    const float* x      = (const float*)d_in[0];
    const float* c      = (const float*)d_in[1];
    const float* w_mod  = (const float*)d_in[2];
    const float* b_mod  = (const float*)d_in[3];
    const float* w_q    = (const float*)d_in[4];
    const float* b_q    = (const float*)d_in[5];
    const float* w_k    = (const float*)d_in[6];
    const float* b_k    = (const float*)d_in[7];
    const float* w_v    = (const float*)d_in[8];
    const float* b_v    = (const float*)d_in[9];
    const float* w_attn = (const float*)d_in[10];
    const float* b_attn = (const float*)d_in[11];
    const float* w_fc1  = (const float*)d_in[12];
    const float* b_fc1  = (const float*)d_in[13];
    const float* w_fc2  = (const float*)d_in[14];
    const float* b_fc2  = (const float*)d_in[15];

    // d_out layout (floats), reference tuple order:
    float* out = (float*)d_out;
    float* o_xout  = out;                                   // [8,1024,1024]
    float* o_xnorm = out + (size_t)MM * DD;                 // [8,1024,1024]
    float* o_xmod  = out + (size_t)2 * MM * DD;             // [8,1024,1024]
    float* o_shift = out + (size_t)3 * MM * DD;             // [8,1024]
    float* o_scale = o_shift + (size_t)BB * DD;             // [8,1024]
    float* o_cmod  = o_scale + (size_t)BB * DD;             // [8,6144]
    float* o_cact  = o_cmod + (size_t)BB * 6 * DD;          // [8,1024]

    // workspace layout (bytes)
    char* ws = (char*)d_ws;
    size_t off = 0;
    auto take = [&](size_t bytes) { char* p = ws + off; off += (bytes + 255) & ~(size_t)255; return p; };
    unsigned short* wq_b   = (unsigned short*)take((size_t)DD * DD * 2);
    unsigned short* wk_b   = (unsigned short*)take((size_t)DD * DD * 2);
    unsigned short* wv_b   = (unsigned short*)take((size_t)DD * DD * 2);
    unsigned short* wat_b  = (unsigned short*)take((size_t)DD * DD * 2);
    unsigned short* wf1_b  = (unsigned short*)take((size_t)DD * FF * 2);
    unsigned short* wf2_b  = (unsigned short*)take((size_t)FF * DD * 2);
    unsigned short* xmod_b = (unsigned short*)take((size_t)MM * DD * 2);
    unsigned short* q_b    = (unsigned short*)take((size_t)MM * DD * 2);
    unsigned short* k_b    = (unsigned short*)take((size_t)MM * DD * 2);
    unsigned short* v_b    = (unsigned short*)take((size_t)MM * DD * 2);
    unsigned short* y_b    = (unsigned short*)take((size_t)MM * DD * 2);
    float*          x1_f   = (float*)take((size_t)MM * DD * 4);
    unsigned short* xm2_b  = (unsigned short*)take((size_t)MM * DD * 2);
    unsigned short* h_b    = (unsigned short*)take((size_t)MM * FF * 2);

    // 1) weights -> bf16
    {
        int nDD = DD * DD, nDF = DD * FF;
        f32_to_bf16_kernel<<<(nDD + 255) / 256, 256, 0, stream>>>(w_q,    wq_b,  nDD);
        f32_to_bf16_kernel<<<(nDD + 255) / 256, 256, 0, stream>>>(w_k,    wk_b,  nDD);
        f32_to_bf16_kernel<<<(nDD + 255) / 256, 256, 0, stream>>>(w_v,    wv_b,  nDD);
        f32_to_bf16_kernel<<<(nDD + 255) / 256, 256, 0, stream>>>(w_attn, wat_b, nDD);
        f32_to_bf16_kernel<<<(nDF + 255) / 256, 256, 0, stream>>>(w_fc1,  wf1_b, nDF);
        f32_to_bf16_kernel<<<(nDF + 255) / 256, 256, 0, stream>>>(w_fc2,  wf2_b, nDF);
    }

    // 2) conditioning path: cact, cmod (+ shift/scale copies)
    silu_kernel<<<(BB * DD + 255) / 256, 256, 0, stream>>>(c, o_cact, BB * DD);
    cmod_kernel<<<dim3(6 * DD / 256, BB), 256, 0, stream>>>(
        o_cact, w_mod, b_mod, o_cmod, o_shift, o_scale);

    // 3) LN1 + modulate (msa): x_norm, x_modulated (f32 outs) + bf16 copy
    ln_mod_kernel<<<MM, 256, 0, stream>>>(x, o_cmod, 0, DD,
                                          o_xnorm, o_xmod, xmod_b);

    // 4) QKV projections (q scaled by 1/Dh)
    dim3 gD(DD / 64, MM / 128);
    gemm_bf16_kernel<0><<<gD, 256, 0, stream>>>(xmod_b, wq_b, b_q, nullptr, 0,
        nullptr, nullptr, q_b, MM, DD, DD, 1.0f / (float)DH);
    gemm_bf16_kernel<0><<<gD, 256, 0, stream>>>(xmod_b, wk_b, b_k, nullptr, 0,
        nullptr, nullptr, k_b, MM, DD, DD, 1.0f);
    gemm_bf16_kernel<0><<<gD, 256, 0, stream>>>(xmod_b, wv_b, b_v, nullptr, 0,
        nullptr, nullptr, v_b, MM, DD, DD, 1.0f);

    // 5) flash attention
    attn_kernel<<<dim3(SS / 128, BB * HH), 256, 0, stream>>>(q_b, k_b, v_b, y_b);

    // 6) attn projection + gated residual: x1 = x + gate_msa*(y@w_attn+b)
    gemm_bf16_kernel<1><<<gD, 256, 0, stream>>>(y_b, wat_b, b_attn,
        o_cmod, 2 * DD, x, x1_f, nullptr, MM, DD, DD, 1.0f);

    // 7) LN2 + modulate (mlp): bf16 only
    ln_mod_kernel<<<MM, 256, 0, stream>>>(x1_f, o_cmod, 3 * DD, 4 * DD,
                                          nullptr, nullptr, xm2_b);

    // 8) fc1 + exact GELU
    gemm_bf16_kernel<2><<<dim3(FF / 64, MM / 128), 256, 0, stream>>>(
        xm2_b, wf1_b, b_fc1, nullptr, 0, nullptr, nullptr, h_b,
        MM, FF, DD, 1.0f);

    // 9) fc2 + gated residual -> x_out
    gemm_bf16_kernel<1><<<gD, 256, 0, stream>>>(h_b, wf2_b, b_fc2,
        o_cmod, 5 * DD, x1_f, o_xout, nullptr, MM, DD, FF, 1.0f);
}
